// MultiHeadAttention_45732811768295
// MI455X (gfx1250) — compile-verified, matrix operand order checked by reading, not run on hardware
//
#include <hip/hip_runtime.h>
#include <hip/hip_bf16.h>

// Problem constants (match reference)
#define BATCH 8
#define SEQ   2048
#define DM    1024
#define HD    64
#define MROWS (BATCH * SEQ)   // 16384
#define LDK   80              // padded LDS row pitch (bf16): 160B rows, 16B aligned

typedef __attribute__((ext_vector_type(16))) __bf16 v16bf;
typedef __attribute__((ext_vector_type(8)))  float  v8f;
typedef __attribute__((ext_vector_type(4)))  int    v4i;

// ---------------- CDNA5 async global->LDS copy (ASYNCcnt path) -------------
#if defined(__has_builtin)
#if __has_builtin(__builtin_amdgcn_global_load_async_to_lds_b128)
#define USE_ASYNC_LDS 1
#endif
#endif
#ifndef USE_ASYNC_LDS
#define USE_ASYNC_LDS 0
#endif

typedef __attribute__((address_space(1))) v4i gv4i;  // global int4
typedef __attribute__((address_space(3))) v4i lv4i;  // LDS int4

__device__ __forceinline__ void async_b128(const __bf16* g, __bf16* l) {
#if USE_ASYNC_LDS
  // global -> LDS, 16 bytes per lane, no VGPR round trip, tracked by ASYNCcnt.
  // Low 32 bits of a generic LDS pointer are the LDS byte offset (aperture rule).
  __builtin_amdgcn_global_load_async_to_lds_b128(
      (gv4i*)(uintptr_t)g, (lv4i*)(unsigned)(uintptr_t)l, 0, 0);
#else
  (void)g; (void)l;
#endif
}

__device__ __forceinline__ void wait_async0() {
#if defined(__has_builtin) && __has_builtin(__builtin_amdgcn_s_wait_asynccnt)
  __builtin_amdgcn_s_wait_asynccnt(0);
#else
  asm volatile("s_wait_asynccnt 0x0" ::: "memory");
#endif
}

// ---------------- WMMA helpers ---------------------------------------------
__device__ __forceinline__ v8f wmma_bf16(v16bf a, v16bf b, v8f c) {
  // D = A(16x32 bf16) * B(32x16 bf16) + C(16x16 f32)
  return __builtin_amdgcn_wmma_f32_16x16x32_bf16(false, a, false, b, (short)0, c,
                                                 false, false);
}

// Load a 16x32 bf16 tile in the CDNA5 16-bit A/B lane layout:
// lane = r + 16*h ; lane reads row r, cols {h*8..h*8+7} and {16+h*8..16+h*8+7}.
__device__ __forceinline__ v16bf ld_tile_bf16(const __bf16* __restrict__ base,
                                              int rstride) {
  const int lane = threadIdx.x & 31;
  const int r = lane & 15, h = lane >> 4;
  const __bf16* p = base + (size_t)r * rstride + h * 8;
  union { v16bf v; uint4 q[2]; } u;
  u.q[0] = *(const uint4*)(p);
  u.q[1] = *(const uint4*)(p + 16);
  return u.v;
}

// Same tile pattern, fp32 source, convert to bf16 in registers.
__device__ __forceinline__ v16bf ld_tile_f32(const float* __restrict__ base,
                                             int rstride) {
  const int lane = threadIdx.x & 31;
  const int r = lane & 15, h = lane >> 4;
  const float* p = base + (size_t)r * rstride + h * 8;
  float4 f0 = *(const float4*)(p);
  float4 f1 = *(const float4*)(p + 4);
  float4 f2 = *(const float4*)(p + 16);
  float4 f3 = *(const float4*)(p + 20);
  v16bf v;
  v[0]  = (__bf16)f0.x; v[1]  = (__bf16)f0.y; v[2]  = (__bf16)f0.z; v[3]  = (__bf16)f0.w;
  v[4]  = (__bf16)f1.x; v[5]  = (__bf16)f1.y; v[6]  = (__bf16)f1.z; v[7]  = (__bf16)f1.w;
  v[8]  = (__bf16)f2.x; v[9]  = (__bf16)f2.y; v[10] = (__bf16)f2.z; v[11] = (__bf16)f2.w;
  v[12] = (__bf16)f3.x; v[13] = (__bf16)f3.y; v[14] = (__bf16)f3.z; v[15] = (__bf16)f3.w;
  return v;
}

// ---- prep: W{q,k,v} [1024x64] f32 -> Wt [64x1024] bf16 (transposed) -------
__global__ __launch_bounds__(256) void prep_weights_kernel(
    const float* __restrict__ Wq, const float* __restrict__ Wk,
    const float* __restrict__ Wv, __bf16* __restrict__ wt) {
  const int g = blockIdx.x * 256 + threadIdx.x;  // 3*65536 threads
  const int w = g >> 16;
  const int rem = g & 65535;
  const int k = rem & 1023;
  const int n = rem >> 10;  // 0..63
  const float* W = (w == 0) ? Wq : (w == 1) ? Wk : Wv;
  wt[(size_t)w * (HD * DM) + (size_t)n * DM + k] = (__bf16)W[(size_t)k * HD + n];
}

// ---- prep: Wo_sum^T [1024x64] bf16, Wo_sum[d][c] = sum_h Wo[h*64+d][c] ----
__global__ __launch_bounds__(256) void prep_wosum_kernel(
    const float* __restrict__ Wo, __bf16* __restrict__ wost) {
  const int g = blockIdx.x * 256 + threadIdx.x;  // 65536 threads
  const int k = g & 63;    // d
  const int n = g >> 6;    // c
  float s = 0.f;
#pragma unroll
  for (int h = 0; h < 16; ++h) s += Wo[(size_t)(h * 64 + k) * DM + n];
  wost[(size_t)n * HD + k] = (__bf16)s;
}

// ---- q/k/v projection: [16384x1024] f32 @ Wt^T -> [16384x64] bf16 ---------
// transpose_out == 0: out[row][d]  (q, k)
// transpose_out == 1: out[d][row]  (v stored transposed; flash consumes V^T tiles)
__global__ __launch_bounds__(128) void proj_kernel(
    const float* __restrict__ X, const __bf16* __restrict__ Wt,
    const float* __restrict__ bias, __bf16* __restrict__ out, float scale,
    int transpose_out) {
  const int wave = threadIdx.x >> 5;
  const int lane = threadIdx.x & 31;
  const int row0 = blockIdx.x * 64 + wave * 16;
  v8f a0 = {}, a1 = {}, a2 = {}, a3 = {};
  for (int kc = 0; kc < DM; kc += 32) {
    const v16bf a = ld_tile_f32(X + (size_t)row0 * DM + kc, DM);
    a0 = wmma_bf16(a, ld_tile_bf16(Wt + (size_t)(0 * 16) * DM + kc, DM), a0);
    a1 = wmma_bf16(a, ld_tile_bf16(Wt + (size_t)(1 * 16) * DM + kc, DM), a1);
    a2 = wmma_bf16(a, ld_tile_bf16(Wt + (size_t)(2 * 16) * DM + kc, DM), a2);
    a3 = wmma_bf16(a, ld_tile_bf16(Wt + (size_t)(3 * 16) * DM + kc, DM), a3);
  }
  const int half = lane >> 4, nl = lane & 15;
  const float b0 = bias[0 * 16 + nl];
  const float b1 = bias[1 * 16 + nl];
  const float b2 = bias[2 * 16 + nl];
  const float b3 = bias[3 * 16 + nl];
  if (!transpose_out) {
#pragma unroll
    for (int j = 0; j < 8; ++j) {
      const int m = j + 8 * half;  // C layout: vgpr j, lane half selects M
      const size_t orow = (size_t)(row0 + m) * HD;
      out[orow + 0 * 16 + nl] = (__bf16)((a0[j] + b0) * scale);
      out[orow + 1 * 16 + nl] = (__bf16)((a1[j] + b1) * scale);
      out[orow + 2 * 16 + nl] = (__bf16)((a2[j] + b2) * scale);
      out[orow + 3 * 16 + nl] = (__bf16)((a3[j] + b3) * scale);
    }
  } else {
#pragma unroll
    for (int j = 0; j < 8; ++j) {
      const int m = j + 8 * half;
      const size_t r = (size_t)(row0 + m);
      out[(size_t)(0 * 16 + nl) * MROWS + r] = (__bf16)((a0[j] + b0) * scale);
      out[(size_t)(1 * 16 + nl) * MROWS + r] = (__bf16)((a1[j] + b1) * scale);
      out[(size_t)(2 * 16 + nl) * MROWS + r] = (__bf16)((a2[j] + b2) * scale);
      out[(size_t)(3 * 16 + nl) * MROWS + r] = (__bf16)((a3[j] + b3) * scale);
    }
  }
}

// ---- flash attention: per (batch, 64 query rows); online softmax ----------
// Double-buffered K/V tiles fetched with GLOBAL_LOAD_ASYNC_TO_LDS_B128 so the
// next tile streams into LDS while WMMA works on the current one.
__global__ __launch_bounds__(128) void flash_kernel(
    const __bf16* __restrict__ qw, const __bf16* __restrict__ kw,
    const __bf16* __restrict__ vt, __bf16* __restrict__ ctx) {
  __shared__ __bf16 Kb[2][64][LDK];    // key-block, row-major [key][d]
  __shared__ __bf16 Vb[2][64][LDK];    // value-block transposed [d][key]
  __shared__ __bf16 Pt[4][16][LDK];    // per-wave probabilities [m][key]
  const int b    = blockIdx.y;
  const int qr0  = blockIdx.x * 64;
  const int tid  = threadIdx.x;
  const int lane = tid & 31;
  const int wave = tid >> 5;
  const int half = lane >> 4;
  const int nl   = lane & 15;
  const int i    = tid >> 1;           // copy row 0..63
  const int hh   = tid & 1;            // copy half-row

  const __bf16* kbase = kw + (size_t)b * SEQ * HD;  // [2048][64] rows
  const __bf16* vbase = vt + (size_t)b * SEQ;       // [64][MROWS], col offset b*SEQ

  const int qrow = b * SEQ + qr0 + wave * 16;
  const v16bf aq0 = ld_tile_bf16(qw + (size_t)qrow * HD, HD);       // d 0..31
  const v16bf aq1 = ld_tile_bf16(qw + (size_t)qrow * HD + 32, HD);  // d 32..63

  v8f o0 = {}, o1 = {}, o2 = {}, o3 = {};
  float rmax[8], rsum[8];
#pragma unroll
  for (int j = 0; j < 8; ++j) { rmax[j] = -3.0e38f; rsum[j] = 0.f; }

#if USE_ASYNC_LDS
  {  // prologue: stream tile kb=0 into buffer 0
    const __bf16* ks = kbase + (size_t)i * HD + hh * 32;
    const __bf16* vs = vbase + (size_t)i * MROWS + hh * 32;
#pragma unroll
    for (int c = 0; c < 4; ++c) {
      async_b128(ks + c * 8, &Kb[0][i][hh * 32 + c * 8]);
      async_b128(vs + c * 8, &Vb[0][i][hh * 32 + c * 8]);
    }
  }
#endif

  for (int kb = 0; kb < SEQ / 64; ++kb) {
#if USE_ASYNC_LDS
    const int cur = kb & 1;
    wait_async0();        // my async writes for buffer `cur` have landed
    __syncthreads();      // everyone's have; prior compute on `cur^1` is done
    if (kb + 1 < SEQ / 64) {  // stream next tile while we compute on this one
      const __bf16* ks = kbase + (size_t)(kb + 1) * 64 * HD + (size_t)i * HD + hh * 32;
      const __bf16* vs = vbase + (size_t)(kb + 1) * 64 + (size_t)i * MROWS + hh * 32;
#pragma unroll
      for (int c = 0; c < 4; ++c) {
        async_b128(ks + c * 8, &Kb[cur ^ 1][i][hh * 32 + c * 8]);
        async_b128(vs + c * 8, &Vb[cur ^ 1][i][hh * 32 + c * 8]);
      }
    }
    __bf16 (*Kt)[LDK] = Kb[cur];
    __bf16 (*Vt)[LDK] = Vb[cur];
#else
    __syncthreads();
    {  // synchronous fallback copy (both tiles are plain row-major 2D copies)
      const uint4* ksrc = (const uint4*)(kbase + (size_t)(kb * 64 + i) * HD + hh * 32);
      uint4* kdst = (uint4*)(&Kb[0][i][hh * 32]);
      kdst[0] = ksrc[0]; kdst[1] = ksrc[1]; kdst[2] = ksrc[2]; kdst[3] = ksrc[3];
      const uint4* vsrc = (const uint4*)(vbase + (size_t)i * MROWS + kb * 64 + hh * 32);
      uint4* vdst = (uint4*)(&Vb[0][i][hh * 32]);
      vdst[0] = vsrc[0]; vdst[1] = vsrc[1]; vdst[2] = vsrc[2]; vdst[3] = vsrc[3];
    }
    __syncthreads();
    __bf16 (*Kt)[LDK] = Kb[0];
    __bf16 (*Vt)[LDK] = Vb[0];
#endif

    // scores S = Q (16x64) @ K^T (64x64) -> 4 C tiles, f32
    v8f s0 = {}, s1 = {}, s2 = {}, s3 = {};
    s0 = wmma_bf16(aq0, ld_tile_bf16(&Kt[0][0],  LDK), s0);
    s0 = wmma_bf16(aq1, ld_tile_bf16(&Kt[0][32], LDK), s0);
    s1 = wmma_bf16(aq0, ld_tile_bf16(&Kt[16][0],  LDK), s1);
    s1 = wmma_bf16(aq1, ld_tile_bf16(&Kt[16][32], LDK), s1);
    s2 = wmma_bf16(aq0, ld_tile_bf16(&Kt[32][0],  LDK), s2);
    s2 = wmma_bf16(aq1, ld_tile_bf16(&Kt[32][32], LDK), s2);
    s3 = wmma_bf16(aq0, ld_tile_bf16(&Kt[48][0],  LDK), s3);
    s3 = wmma_bf16(aq1, ld_tile_bf16(&Kt[48][32], LDK), s3);

    // online softmax: row = j + 8*half lives across the 16 lanes of this half
#pragma unroll
    for (int j = 0; j < 8; ++j) {
      float m = fmaxf(fmaxf(s0[j], s1[j]), fmaxf(s2[j], s3[j]));
      m = fmaxf(m, __shfl_xor(m, 1, 32));
      m = fmaxf(m, __shfl_xor(m, 2, 32));
      m = fmaxf(m, __shfl_xor(m, 4, 32));
      m = fmaxf(m, __shfl_xor(m, 8, 32));
      m = fmaxf(m, rmax[j]);
      const float scal = __expf(rmax[j] - m);
      rmax[j] = m;
      const float p0 = __expf(s0[j] - m);
      const float p1 = __expf(s1[j] - m);
      const float p2 = __expf(s2[j] - m);
      const float p3 = __expf(s3[j] - m);
      float l = p0 + p1 + p2 + p3;
      l += __shfl_xor(l, 1, 32);
      l += __shfl_xor(l, 2, 32);
      l += __shfl_xor(l, 4, 32);
      l += __shfl_xor(l, 8, 32);
      rsum[j] = rsum[j] * scal + l;
      o0[j] *= scal; o1[j] *= scal; o2[j] *= scal; o3[j] *= scal;
      const int mr = j + 8 * half;
      Pt[wave][mr][0 * 16 + nl] = (__bf16)p0;
      Pt[wave][mr][1 * 16 + nl] = (__bf16)p1;
      Pt[wave][mr][2 * 16 + nl] = (__bf16)p2;
      Pt[wave][mr][3 * 16 + nl] = (__bf16)p3;
    }

    // ctx += P (16x64) @ V (64x64): re-read P in A layout (same-wave DS
    // ordering; compiler inserts the dscnt waits)
    const v16bf ap0 = ld_tile_bf16(&Pt[wave][0][0],  LDK);
    const v16bf ap1 = ld_tile_bf16(&Pt[wave][0][32], LDK);
    o0 = wmma_bf16(ap0, ld_tile_bf16(&Vt[0][0],   LDK), o0);
    o0 = wmma_bf16(ap1, ld_tile_bf16(&Vt[0][32],  LDK), o0);
    o1 = wmma_bf16(ap0, ld_tile_bf16(&Vt[16][0],  LDK), o1);
    o1 = wmma_bf16(ap1, ld_tile_bf16(&Vt[16][32], LDK), o1);
    o2 = wmma_bf16(ap0, ld_tile_bf16(&Vt[32][0],  LDK), o2);
    o2 = wmma_bf16(ap1, ld_tile_bf16(&Vt[32][32], LDK), o2);
    o3 = wmma_bf16(ap0, ld_tile_bf16(&Vt[48][0],  LDK), o3);
    o3 = wmma_bf16(ap1, ld_tile_bf16(&Vt[48][32], LDK), o3);
  }

#pragma unroll
  for (int j = 0; j < 8; ++j) {
    const float inv = 1.f / rsum[j];
    const int mr = j + 8 * half;
    const size_t orow = (size_t)(qrow + mr) * HD;
    ctx[orow + 0 * 16 + nl] = (__bf16)(o0[j] * inv);
    ctx[orow + 1 * 16 + nl] = (__bf16)(o1[j] * inv);
    ctx[orow + 2 * 16 + nl] = (__bf16)(o2[j] * inv);
    ctx[orow + 3 * 16 + nl] = (__bf16)(o3[j] * inv);
  }
}

// ---- output: [16384x64] bf16 @ Wo_sum (via Wo_sum^T) + bo -> f32 ----------
__global__ __launch_bounds__(128) void out_gemm_kernel(
    const __bf16* __restrict__ ctx, const __bf16* __restrict__ wost,
    const float* __restrict__ bo, float* __restrict__ out) {
  const int wave = threadIdx.x >> 5, lane = threadIdx.x & 31;
  const int row0 = blockIdx.x * 64 + wave * 16;
  const int col0 = blockIdx.y * 128;
  v8f acc[8];
#pragma unroll
  for (int t = 0; t < 8; ++t) { v8f z = {}; acc[t] = z; }
#pragma unroll
  for (int ks = 0; ks < 2; ++ks) {
    const v16bf a = ld_tile_bf16(ctx + (size_t)row0 * HD + ks * 32, HD);
#pragma unroll
    for (int t = 0; t < 8; ++t)
      acc[t] = wmma_bf16(
          a, ld_tile_bf16(wost + (size_t)(col0 + t * 16) * HD + ks * 32, HD),
          acc[t]);
  }
  const int half = lane >> 4, nl = lane & 15;
#pragma unroll
  for (int t = 0; t < 8; ++t) {
    const int c = col0 + t * 16 + nl;
    const float bb = bo[c];
#pragma unroll
    for (int j = 0; j < 8; ++j) {
      const int m = j + 8 * half;
      out[(size_t)(row0 + m) * DM + c] = acc[t][j] + bb;
    }
  }
}

extern "C" void kernel_launch(void* const* d_in, const int* in_sizes, int n_in,
                              void* d_out, int out_size, void* d_ws, size_t ws_size,
                              hipStream_t stream) {
  (void)in_sizes; (void)n_in; (void)out_size; (void)ws_size;
  const float* query = (const float*)d_in[0];
  const float* key_  = (const float*)d_in[1];
  const float* value = (const float*)d_in[2];
  const float* Wq = (const float*)d_in[3];
  const float* bq = (const float*)d_in[4];
  const float* Wk = (const float*)d_in[5];
  const float* bk = (const float*)d_in[6];
  const float* Wv = (const float*)d_in[7];
  const float* bv = (const float*)d_in[8];
  const float* Wo = (const float*)d_in[9];
  const float* bo = (const float*)d_in[10];
  float* out = (float*)d_out;

  // workspace layout (all bf16, ~8.5 MB total; L2-resident intermediates)
  char* ws = (char*)d_ws;
  __bf16* wt   = (__bf16*)(ws);                           // 3 * 64*1024
  __bf16* wost = (__bf16*)(ws + 0x60000);                 // 1024*64
  __bf16* qw   = (__bf16*)(ws + 0x80000);                 // 16384*64
  __bf16* kw   = qw + (size_t)MROWS * HD;                 // 16384*64
  __bf16* vtw  = kw + (size_t)MROWS * HD;                 // 64*16384 (transposed)
  __bf16* ctx  = vtw + (size_t)MROWS * HD;                // 16384*64

  prep_weights_kernel<<<768, 256, 0, stream>>>(Wq, Wk, Wv, wt);
  prep_wosum_kernel<<<256, 256, 0, stream>>>(Wo, wost);
  // fold 1/sqrt(head_dim)=1/8 into q (bias included, matches reference math)
  proj_kernel<<<MROWS / 64, 128, 0, stream>>>(query, wt,                       bq, qw,  0.125f, 0);
  proj_kernel<<<MROWS / 64, 128, 0, stream>>>(key_,  wt + (size_t)HD * DM,     bk, kw,  1.0f,   0);
  proj_kernel<<<MROWS / 64, 128, 0, stream>>>(value, wt + (size_t)2 * HD * DM, bv, vtw, 1.0f,   1);
  flash_kernel<<<dim3(SEQ / 64, BATCH), 128, 0, stream>>>(qw, kw, vtw, ctx);
  out_gemm_kernel<<<dim3(MROWS / 64, DM / 128), 128, 0, stream>>>(ctx, wost, bo, out);
}
